// GCNSeq2Seq_89395449299165
// MI455X (gfx1250) — matrix operand order repaired and checked by reading, not compiled, not run on hardware
//
#include <hip/hip_runtime.h>

// ---------------- problem constants (from reference) ----------------
#define Bsz    8192
#define T_IN_  14
#define F_IN_  4
#define HID_   64
#define F_OUT_ 2
#define T_OUT_ 106
#define Nn     (Bsz * T_IN_)   // 114688 nodes
#define Ee     (16 * Nn)       // 1835008 edges

typedef __attribute__((ext_vector_type(16))) _Float16 v16h;
typedef __attribute__((ext_vector_type(8)))  _Float16 v8h;
typedef __attribute__((ext_vector_type(2)))  _Float16 v2h;
typedef __attribute__((ext_vector_type(8)))  float    v8f;

// WMMA 16x16x32 f16 layouts (CDNA5 ISA §7.12.2/7.12.4), lane half = lane>>4:
//   A (16x32 MxK), row m = lane&15:  j 0..7 -> K = 8*half + j ; j 8..15 -> K = 16 + 8*half + (j-8)
//     => two contiguous 8xf16 (16 B) groups at f16-offsets {8*half, 16+8*half}
//   B (32x16 KxN), col n = lane&15:  j 0..7 -> K = 8*half + j ; j 8..15 -> K = 16 + 8*half + (j-8)
//     => two contiguous 8xf16 (16 B) groups at f16-offsets {8*half, 16+8*half}
//   C/D: VGPR r -> M = r + 8*half, N = lane&15

__device__ __forceinline__ v16h cat8(v8h lo, v8h hi) {
    return __builtin_shufflevector(lo, hi, 0,1,2,3,4,5,6,7,8,9,10,11,12,13,14,15);
}

// ---------------- degree / normalization ----------------
__global__ void k_deg_init(float* __restrict__ deg) {
    int i = blockIdx.x * 256 + threadIdx.x;
    if (i < Nn) deg[i] = 1.0f;                    // self-loop contribution
}

__global__ void k_deg_count(const long long* __restrict__ ei, float* __restrict__ deg) {
    int e = blockIdx.x * 256 + threadIdx.x;
    if (e < Ee) atomicAdd(&deg[(int)ei[Ee + e]], 1.0f);   // col = target
}

__global__ void k_rsqrt(float* __restrict__ deg) {
    int i = blockIdx.x * 256 + threadIdx.x;
    if (i < Nn) deg[i] = rsqrtf(deg[i]);          // deg >= 1 always (self-loops)
}

// ---------------- layer 1: g1 = xf@W1 ; acc1 init = self-loop term ----------------
__global__ void k_g1(const float* __restrict__ xf, const float* __restrict__ W1,
                     const float* __restrict__ dis,
                     float* __restrict__ g1, float* __restrict__ acc1) {
    int idx = blockIdx.x * 256 + threadIdx.x;
    if (idx >= Nn * HID_) return;
    int n = idx >> 6, h = idx & 63;
    const float* xr = xf + n * F_IN_;
    float g = xr[0] * W1[h] + xr[1] * W1[64 + h] + xr[2] * W1[128 + h] + xr[3] * W1[192 + h];
    g1[idx] = g;
    float d = dis[n];
    acc1[idx] = g * d * d;                        // self-loop: norm = dis[i]^2
}

// ---------------- edge scatter: wave per edge, lane handles 2 features ----------------
__global__ void k_scatter(const long long* __restrict__ ei, const float* __restrict__ dis,
                          const float* __restrict__ g, float* __restrict__ acc) {
    int wave = (blockIdx.x * 256 + (int)threadIdx.x) >> 5;
    int lane = threadIdx.x & 31;
    if (wave >= Ee) return;
    int r = (int)ei[wave];         // source
    int c = (int)ei[Ee + wave];    // target
    float nrm = dis[r] * dis[c];
    float2 v = *(const float2*)(g + r * HID_ + lane * 2);   // coalesced 256 B / wave
    atomicAdd(&acc[c * HID_ + lane * 2 + 0], v.x * nrm);    // fp32 atomics resolve in L2
    atomicAdd(&acc[c * HID_ + lane * 2 + 1], v.y * nrm);
}

// ---------------- h1 = relu(acc1 + b1), packed to f16 (feeds WMMA GEMM) ----------------
__global__ void k_relu_f16(const float* __restrict__ acc1, const float* __restrict__ b1,
                           v2h* __restrict__ h1f) {
    int idx = blockIdx.x * 256 + threadIdx.x;     // over Nn*32 f16-pairs
    if (idx >= Nn * 32) return;
    int h = (idx & 31) * 2;
    float2 v = *(const float2*)(acc1 + (size_t)idx * 2);
    v2h o;
    o[0] = (_Float16)fmaxf(v.x + b1[h + 0], 0.0f);
    o[1] = (_Float16)fmaxf(v.y + b1[h + 1], 0.0f);
    h1f[idx] = o;
}

// ------- layer 2 GEMM: g2 = h1 @ W2 (114688x64 . 64x64) via WMMA, fused acc2 init -------
__global__ __launch_bounds__(256) void k_gemm_h1w2(const _Float16* __restrict__ h1f,
                                                   const float* __restrict__ W2,
                                                   const float* __restrict__ dis,
                                                   float* __restrict__ g2,
                                                   float* __restrict__ acc2) {
    // W2 transposed + f16 in LDS: row n holds column n of W2 contiguously in K.
    // Row stride 72 f16 = 144 B (16B-aligned; 36-dword bank stride -> 16 lanes, 16 banks).
    __shared__ __align__(16) _Float16 W2t[HID_][72];
    int tid = threadIdx.x;
    for (int i = tid; i < HID_ * HID_; i += 256) {
        int n = i & 63, k = i >> 6;
        W2t[n][k] = (_Float16)W2[k * HID_ + n];
    }
    __syncthreads();

    // grid sized exactly: 8 waves/block x 3584 blocks = 28672 tiles = (Nn/16)*4
    int gwave = (blockIdx.x * 256 + tid) >> 5;
    int lane  = tid & 31;
    int tm = gwave >> 2, tn = gwave & 3;
    int m  = tm * 16 + (lane & 15);
    int n  = tn * 16 + (lane & 15);
    int half8  = ((lane >> 4) & 1) * 8;
    int half16 = half8 * 2;

    const _Float16* arow = h1f + (size_t)m * HID_;
    const _Float16* brow = &W2t[n][0];
    v8f c = {};
    #pragma unroll
    for (int kb = 0; kb < HID_; kb += 32) {
        v16h a = cat8(*(const v8h*)(arow + kb + half8),        // global_load_b128 x2
                      *(const v8h*)(arow + kb + 16 + half8));
        v16h b = cat8(*(const v8h*)(brow + kb + half8),        // ds_load_b128 x2
                      *(const v8h*)(brow + kb + 16 + half8));
        c = __builtin_amdgcn_wmma_f32_16x16x32_f16(false, a, false, b,
                                                   (short)0, c, false, false);
    }
    int mb = tm * 16 + ((lane >> 4) ? 8 : 0);
    #pragma unroll
    for (int r = 0; r < 8; ++r) {
        int mr = mb + r;
        float d = dis[mr];
        g2[mr * HID_ + n]   = c[r];               // gathered by scatter pass
        acc2[mr * HID_ + n] = c[r] * d * d;       // fused self-loop init (k_init2 removed)
    }
}

// ---------------- per-batch: (64x14)@Wt(14x106)+bt, then @Wo(64x2)+bo ----------------
__global__ __launch_bounds__(256) void k_final(const float* __restrict__ acc2,
                                               const float* __restrict__ b2,
                                               const float* __restrict__ Wt,
                                               const float* __restrict__ bt,
                                               const float* __restrict__ Wo,
                                               const float* __restrict__ bo,
                                               float* __restrict__ out) {
    __shared__ __align__(16) _Float16 As[HID_][40];   // h2^T per batch, K 14->32, 80 B rows
    __shared__ __align__(16) _Float16 Bst[112][40];   // Wt^T: row=t_out, K 14->32, 80 B rows
    __shared__ float Ts[HID_][112];                   // temporal-FC result (28 KB)
    int b = blockIdx.x, tid = threadIdx.x;

    for (int i = tid; i < HID_ * 32; i += 256) {      // stage A (+b2 fused, zero-pad K)
        int hh = i >> 5, t = i & 31;
        float v = (t < T_IN_) ? (acc2[(b * T_IN_ + t) * HID_ + hh] + b2[hh]) : 0.0f;
        As[hh][t] = (_Float16)v;
    }
    for (int i = tid; i < 112 * 32; i += 256) {       // stage B^T (zero-pad K and N)
        int to = i >> 5, t = i & 31;
        Bst[to][t] = (_Float16)((t < T_IN_ && to < T_OUT_) ? Wt[t * T_OUT_ + to] : 0.0f);
    }
    __syncthreads();

    int wave = tid >> 5, lane = tid & 31;
    int half8  = ((lane >> 4) & 1) * 8;
    int half16 = half8 * 2;
    for (int tile = wave; tile < 4 * 7; tile += 8) {  // wave-uniform tile loop
        int tm = tile / 7, tn = tile % 7;
        const _Float16* arow = &As[tm * 16 + (lane & 15)][0];
        const _Float16* brow = &Bst[tn * 16 + (lane & 15)][0];
        v16h a = cat8(*(const v8h*)(arow + half8), *(const v8h*)(arow + 16 + half8));
        v16h bf = cat8(*(const v8h*)(brow + half8), *(const v8h*)(brow + 16 + half8));
        v8f c = {};
        c = __builtin_amdgcn_wmma_f32_16x16x32_f16(false, a, false, bf,
                                                   (short)0, c, false, false);
        int mb = tm * 16 + ((lane >> 4) ? 8 : 0);
        int n  = tn * 16 + (lane & 15);
        #pragma unroll
        for (int r = 0; r < 8; ++r) Ts[mb + r][n] = c[r];
    }
    __syncthreads();

    if (tid < T_OUT_ * F_OUT_) {                      // fused output FC (+bt inside sum)
        int to = tid >> 1, f = tid & 1;
        float btv = bt[to], acc = 0.0f;
        #pragma unroll 8
        for (int hh = 0; hh < HID_; ++hh)
            acc += (Ts[hh][to] + btv) * Wo[hh * F_OUT_ + f];
        out[(b * T_OUT_ + to) * F_OUT_ + f] = acc + bo[f];
    }
}

// ---------------- host entry ----------------
extern "C" void kernel_launch(void* const* d_in, const int* in_sizes, int n_in,
                              void* d_out, int out_size, void* d_ws, size_t ws_size,
                              hipStream_t stream) {
    const float*     x  = (const float*)d_in[0];
    const long long* ei = (const long long*)d_in[1];   // jnp.int64 edge_index [2,E]
    const float*     W1 = (const float*)d_in[2];
    const float*     b1 = (const float*)d_in[3];
    const float*     W2 = (const float*)d_in[4];
    const float*     b2 = (const float*)d_in[5];
    const float*     Wt = (const float*)d_in[6];
    const float*     bt = (const float*)d_in[7];
    const float*     Wo = (const float*)d_in[8];
    const float*     bo = (const float*)d_in[9];
    float* out = (float*)d_out;

    // workspace: dis (N f32) | gbuf (N*64 f32) | accbuf (N*64 f32) | h1f (N*64 f16) ~ 74 MB
    size_t floats = (size_t)Nn + 2ull * Nn * HID_;
    size_t need   = floats * sizeof(float) + (size_t)Nn * HID_ * sizeof(_Float16);
    if (ws_size < need) return;
    float*    deg  = (float*)d_ws;
    float*    gbuf = deg + Nn;
    float*    acc  = gbuf + (size_t)Nn * HID_;
    _Float16* h1f  = (_Float16*)(acc + (size_t)Nn * HID_);  // 16B-aligned (prefix % 16 == 0)

    const int NB_N  = (Nn + 255) / 256;            // 448
    const int NB_E  = (Ee + 255) / 256;            // 7168
    const int NB_NH = (Nn * HID_ + 255) / 256;     // 28672
    const int NB_RL = (Nn * 32 + 255) / 256;       // 14336
    const int NB_SC = Ee / 8;                      // 229376 (8 waves/block, 1 edge/wave)
    const int NB_GM = (Nn / 16 * 4) / 8;           // 3584   (28672 tiles, exact fit)

    k_deg_init <<<NB_N,  256, 0, stream>>>(deg);
    k_deg_count<<<NB_E,  256, 0, stream>>>(ei, deg);
    k_rsqrt    <<<NB_N,  256, 0, stream>>>(deg);                        // deg -> dis in place
    k_g1       <<<NB_NH, 256, 0, stream>>>(x, W1, deg, gbuf, acc);      // g1 + self-loop init
    k_scatter  <<<NB_SC, 256, 0, stream>>>(ei, deg, gbuf, acc);         // layer-1 message pass
    k_relu_f16 <<<NB_RL, 256, 0, stream>>>(acc, b1, (v2h*)h1f);         // h1 = relu(.+b1), f16
    k_gemm_h1w2<<<NB_GM, 256, 0, stream>>>(h1f, W2, deg, gbuf, acc);    // g2 (WMMA) + acc2 init
    k_scatter  <<<NB_SC, 256, 0, stream>>>(ei, deg, gbuf, acc);         // layer-2 message pass
    k_final    <<<Bsz,   256, 0, stream>>>(acc, b2, Wt, bt, Wo, bo, out); // temporal + out FC
}
